// IIRFilterConv_65850438582774
// MI455X (gfx1250) — matrix operand order repaired and checked by reading, not compiled, not run on hardware
//
#include <hip/hip_runtime.h>
#include <hip/hip_fp16.h>

typedef __attribute__((ext_vector_type(16))) _Float16 v16h;
typedef __attribute__((ext_vector_type(8)))  float    v8f;

#define N_FILT   80
#define FILT_DIM 250
#define K_PAD    256
#define FS_F     16000.0f
#define L_IN     32000
#define L_OUT    (L_IN - FILT_DIM + 1)   // 31751
#define M_TILES  5                        // 80 / 16
#define KB       8                        // 256 / 32
#define TN       128                      // output columns per workgroup
#define STAGE    384                      // TN + K_PAD (staged signal halves)

#ifndef M_PI_F
#define M_PI_F 3.14159265358979323846f
#endif

// ---------------------------------------------------------------------------
// Kernel 1: build the 80x250 SincNet IIR filters in fp32, normalize, window,
// and write f16 taps PRE-SWIZZLED into the CDNA5 16-bit A-matrix (16x32)
// fragment layout so the GEMM kernel loads A with plain b128 loads.
// Apack layout: [m_tile][kb][lane][16 halves]  (5*8*32*16 = 20480 halves)
// ---------------------------------------------------------------------------
__global__ __launch_bounds__(256) void build_filters_kernel(
    const float* __restrict__ nf1, const float* __restrict__ nf2,
    _Float16* __restrict__ apack) {
  __shared__ float red[256];
  __shared__ float s_min, s_max, s_mean;

  const int f = blockIdx.x;     // filter 0..79
  const int i = threadIdx.x;    // tap 0..255 (250..255 are zero pad)
  const bool valid = (i < FILT_DIM);

  // band edges (same math as reference, fp32)
  float f1 = fminf(fmaxf(fabsf(nf1[f]), 0.0f), 0.5f);
  float f2 = fminf(f1 + fabsf(nf2[f] - f1), 0.5f);
  float f1hz = f1 * FS_F, f2hz = f2 * FS_F;
  float fc = 0.5f * (f1hz + f2hz);
  float bw = f2hz - f1hz;

  // damped sinusoid tap
  float h = 0.0f;
  if (valid) {
    int j = (i < 125) ? (124 - i) : (i - 125);          // flip for left half
    float lin = 125.0f * (float)j / 124.0f;             // linspace(0,125,125)
    float t = ((i < 125) ? (1.0f + lin) : lin) / FS_F;  // t_right1 / t_right
    h = __expf(-bw * t) * __sinf(2.0f * M_PI_F * fc * t);
  }

  // --- min reduction ---
  red[i] = valid ? h : __builtin_inff();
  __syncthreads();
  for (int s = 128; s > 0; s >>= 1) {
    if (i < s) red[i] = fminf(red[i], red[i + s]);
    __syncthreads();
  }
  if (i == 0) s_min = red[0];
  __syncthreads();

  // --- max reduction ---
  red[i] = valid ? h : -__builtin_inff();
  __syncthreads();
  for (int s = 128; s > 0; s >>= 1) {
    if (i < s) red[i] = fmaxf(red[i], red[i + s]);
    __syncthreads();
  }
  if (i == 0) s_max = red[0];
  __syncthreads();

  // normalize to [-1,1]
  float hn = valid ? (2.0f * (h - s_min) / (s_max - s_min + 1e-6f) - 1.0f) : 0.0f;

  // --- mean reduction (of normalized taps) ---
  red[i] = hn;
  __syncthreads();
  for (int s = 128; s > 0; s >>= 1) {
    if (i < s) red[i] = red[i] + red[i + s];
    __syncthreads();
  }
  if (i == 0) s_mean = red[0] / (float)FILT_DIM;
  __syncthreads();

  // Hamming window: n = linspace(0,250,250) -> win(i) = 0.54-0.46*cos(2*pi*i/249)
  float win = 0.54f - 0.46f * __cosf(2.0f * M_PI_F * (float)i / 249.0f);
  float val = valid ? (hn - s_mean) * win : 0.0f;

  // swizzle into A-fragment layout: k = i
  const int m  = f >> 4;          // M tile
  const int M  = f & 15;          // row in tile
  const int kb = i >> 5;          // 32-wide K block
  const int r  = i & 31;
  const int g  = r >> 4;                       // lane group (0: lanes 0-15)
  const int lane = M + 16 * g;
  const int hh = r & 15;                       // half index within v16h
  apack[(((m * KB + kb) * 32) + lane) * 16 + hh] = (_Float16)val;
}

// ---------------------------------------------------------------------------
// Kernel 2: Toeplitz GEMM with v_wmma_f32_16x16x32_f16.
// Block = 256 threads (8 waves). Block covers (batch b, 128 output cols).
// Stage signal tile (f32 -> f16) into LDS once; each wave owns 16 columns
// and loops over 5 M-tiles x 8 K-blocks of WMMA with f32 accumulation.
// ---------------------------------------------------------------------------
__global__ __launch_bounds__(256) void sinc_conv_wmma_kernel(
    const float* __restrict__ x, const _Float16* __restrict__ apack,
    float* __restrict__ out, int n_batch) {
  __shared__ _Float16 s_sig[STAGE];

  const int b  = blockIdx.y;
  const int t0 = blockIdx.x * TN;
  const int tid = threadIdx.x;
  const float* xb = x + (size_t)b * L_IN;

  // stage 384 samples, convert f32 -> f16, zero-fill past end of signal
  for (int idx = tid; idx < STAGE; idx += 256) {
    int g = t0 + idx;
    float v = (g < L_IN) ? xb[g] : 0.0f;
    s_sig[idx] = (_Float16)v;
  }
  __syncthreads();

  const int wave = tid >> 5;
  const int lane = tid & 31;
  const int n    = lane & 15;     // column within 16-wide subtile
  const int grp  = lane >> 4;     // B/C lane group
  const int n0   = wave * 16;     // this wave's local column base
  const int col  = t0 + n0 + n;

  const v16h* __restrict__ ap = (const v16h*)apack;

  union U16 { v16h v; _Float16 e[16]; };
  union U8  { v8f  v; float    e[8];  };

  for (int m = 0; m < M_TILES; ++m) {
    U8 acc;
    acc.v = (v8f){};
#pragma unroll
    for (int kb = 0; kb < KB; ++kb) {
      // A fragment: pre-swizzled, 32B contiguous per lane (2x b128)
      v16h afrag = ap[(m * KB + kb) * 32 + lane];
      // B fragment: lane group 0 holds K = kb*32+0..15, group 1 holds +16..31
      U16 bf;
      const int base = n0 + n + kb * 32 + grp * 16;
#pragma unroll
      for (int j = 0; j < 16; ++j) bf.e[j] = s_sig[base + j];
      acc.v = __builtin_amdgcn_wmma_f32_16x16x32_f16(
          false, afrag, false, bf.v, (short)0, acc.v, false, false);
    }
    // C layout: VGPR r -> row m*16 + r (lanes 0-15) / m*16 + 8 + r (lanes 16-31)
    if (col < L_OUT) {
      const int rowbase = m * 16 + grp * 8;
      const size_t obase = ((size_t)b * N_FILT + rowbase) * (size_t)L_OUT + col;
#pragma unroll
      for (int r = 0; r < 8; ++r)
        __builtin_nontemporal_store(acc.e[r], out + obase + (size_t)r * L_OUT);
    }
  }
  (void)n_batch;
}

// ---------------------------------------------------------------------------
extern "C" void kernel_launch(void* const* d_in, const int* in_sizes, int n_in,
                              void* d_out, int out_size, void* d_ws, size_t ws_size,
                              hipStream_t stream) {
  const float* x   = (const float*)d_in[0];
  const float* nf1 = (const float*)d_in[1];
  const float* nf2 = (const float*)d_in[2];
  float* out = (float*)d_out;

  _Float16* apack = (_Float16*)d_ws;   // 40 KB packed filter bank

  const int n_batch = in_sizes[0] / L_IN;   // 32

  build_filters_kernel<<<dim3(N_FILT), dim3(256), 0, stream>>>(nf1, nf2, apack);

  dim3 grid((L_OUT + TN - 1) / TN, n_batch);
  sinc_conv_wmma_kernel<<<grid, dim3(256), 0, stream>>>(x, apack, out, n_batch);

  (void)n_in; (void)out_size; (void)ws_size;
}
